// MultiHeadSelfAttention_23837068493139
// MI455X (gfx1250) — matrix operand order repaired and checked by reading, not compiled, not run on hardware
//
#include <hip/hip_runtime.h>
#include <hip/hip_bf16.h>

typedef __attribute__((ext_vector_type(16))) __bf16 v16bf;
typedef __attribute__((ext_vector_type(8)))  __bf16 v8bf;
typedef __attribute__((ext_vector_type(4)))  __bf16 v4bf;
typedef __attribute__((ext_vector_type(8)))  float  v8f;
typedef __attribute__((ext_vector_type(4)))  int    v4i;

#define L_SEQ  2048
#define NBATCH 2
#define EMB    2048
#define NH     16
#define DH     128
#define MROWS  (L_SEQ * NBATCH)   // 4096 GEMM rows (row = l*N + n)

// ------------------------------------------------ async global->LDS (gfx1250)
#if __has_builtin(__builtin_amdgcn_global_load_async_to_lds_b128) && \
    __has_builtin(__builtin_amdgcn_s_wait_asynccnt)
#define HAS_ASYNC_LDS 1
#else
#define HAS_ASYNC_LDS 0
#endif

// copy 16B from global to LDS (per-lane). Falls back to a sync copy when the
// async builtins are unavailable on this toolchain.
static __device__ __forceinline__ void copy16_g2l(const void* g, void* l) {
#if HAS_ASYNC_LDS
  __builtin_amdgcn_global_load_async_to_lds_b128(
      (__attribute__((address_space(1))) v4i*)g,
      (__attribute__((address_space(3))) v4i*)l, 0, 0);
#else
  *(uint4*)l = *(const uint4*)g;
#endif
}

// drain this wave's outstanding async LDS writes (call before barrier)
static __device__ __forceinline__ void async_join() {
#if HAS_ASYNC_LDS
  __builtin_amdgcn_s_wait_asynccnt(0);
#endif
}

// ---------------------------------------------------------------- WMMA helpers

static __device__ __forceinline__ v8f wmma_bf16(v16bf a, v16bf b, v8f c) {
  return __builtin_amdgcn_wmma_f32_16x16x32_bf16(
      /*neg_a=*/false, a, /*neg_b=*/false, b,
      /*c_mod=*/(short)0, c, /*reuse_a=*/false, /*reuse_b=*/false);
}

// A fragment (16x32 bf16): lane holds row = lane&15; 8 contiguous K at
// k0 = (lane>=16 ? 8 : 0) and 8 more at k0+16.
static __device__ __forceinline__ v16bf frag_a(const __bf16* base, int stride) {
  const int lane = threadIdx.x & 31;
  const __bf16* p = base + (lane & 15) * stride + ((lane >> 4) << 3);
  v8bf lo = *(const v8bf*)(p);
  v8bf hi = *(const v8bf*)(p + 16);
  return __builtin_shufflevector(lo, hi, 0,1,2,3,4,5,6,7,8,9,10,11,12,13,14,15);
}

// B fragment (32x16 bf16) from a row-major [Ncol x K] buffer: lane holds
// col = lane&15; 8 contiguous K at k0 = (lane>=16 ? 8 : 0) and 8 more at k0+16.
static __device__ __forceinline__ v16bf frag_b(const __bf16* base, int stride) {
  const int lane = threadIdx.x & 31;
  const __bf16* p = base + (lane & 15) * stride + ((lane >> 4) << 3);
  v8bf lo = *(const v8bf*)(p);
  v8bf hi = *(const v8bf*)(p + 16);
  return __builtin_shufflevector(lo, hi, 0,1,2,3,4,5,6,7,8,9,10,11,12,13,14,15);
}

// ---------------------------------------------------------------- fp32 -> bf16

__global__ __launch_bounds__(256)
void cvt_f32_bf16(const float* __restrict__ in, __bf16* __restrict__ out, int n4) {
  int i = blockIdx.x * blockDim.x + threadIdx.x;
  if (i >= n4) return;
  float4 v = ((const float4*)in)[i];
  v4bf o;
  o[0] = (__bf16)v.x; o[1] = (__bf16)v.y; o[2] = (__bf16)v.z; o[3] = (__bf16)v.w;
  ((v4bf*)out)[i] = o;
}

// ---------------------------------------------------------------- GEMM + epilogue
// C[m, c] = sum_k A[m,k] * Bt[c,k]  (+ bias[c]), M=4096, Ncol=2048, K=2048.
// 128x128 block tile, 4 waves, each wave a 64x64 (4x4 frag) sub-tile:
// 16 WMMAs per 16 ds_load_b128 per K-step.
// mode 0: Q  -> (v+bias)*scale, layout [N,H,L,D]
// mode 1: K  ->  v+bias,        layout [N,H,L,D]
// mode 2: V  ->  v+bias,        layout [N,H,D,L]  (transposed for PV GEMM)
// mode 3: out->  v+bias, fp32,  layout [M, E] (== [L,N,E] flat)
__global__ __launch_bounds__(128)
void gemm_wmma(const __bf16* __restrict__ A, const __bf16* __restrict__ Bt,
               const float* __restrict__ bias, __bf16* __restrict__ outb,
               float* __restrict__ outf, int mode) {
  constexpr int KD  = EMB;
  constexpr int AST = 40;             // 32 + 8 pad elements (80B rows)
  __shared__ __bf16 As[128 * AST];
  __shared__ __bf16 Bs[128 * AST];

  const int wave = threadIdx.x >> 5;
  const int lane = threadIdx.x & 31;
  const int rowBase = blockIdx.y * 128;
  const int colBase = blockIdx.x * 128;
  const int wr = (wave >> 1) * 64;    // wave's 64x64 sub-tile
  const int wc = (wave & 1) * 64;

  v8f acc[4][4] = {};

  for (int k0 = 0; k0 < KD; k0 += 32) {
    __syncthreads();
    // 128x32 bf16 tiles = 512 x 16B segments each, 4 per thread
    #pragma unroll
    for (int t = 0; t < 4; ++t) {
      int s  = threadIdx.x + t * 128;
      int r  = s >> 2;
      int cs = (s & 3) * 8;
      copy16_g2l(&A [(size_t)(rowBase + r) * KD + k0 + cs], &As[r * AST + cs]);
      copy16_g2l(&Bt[(size_t)(colBase + r) * KD + k0 + cs], &Bs[r * AST + cs]);
    }
    async_join();
    __syncthreads();

    v16bf a[4], b[4];
    #pragma unroll
    for (int i = 0; i < 4; ++i) a[i] = frag_a(&As[(wr + i * 16) * AST], AST);
    #pragma unroll
    for (int j = 0; j < 4; ++j) b[j] = frag_b(&Bs[(wc + j * 16) * AST], AST);
    #pragma unroll
    for (int i = 0; i < 4; ++i)
      #pragma unroll
      for (int j = 0; j < 4; ++j)
        acc[i][j] = wmma_bf16(a[i], b[j], acc[i][j]);
  }

  const float scale = 0.08838834764831845f;   // 1/sqrt(128)
  #pragma unroll
  for (int i = 0; i < 4; ++i) {
    #pragma unroll
    for (int j = 0; j < 4; ++j) {
      const int col  = colBase + wc + j * 16 + (lane & 15);
      const int row0 = rowBase + wr + i * 16 + ((lane >> 4) << 3);
      const float bv = bias[col];
      #pragma unroll
      for (int r = 0; r < 8; ++r) {
        const int row = row0 + r;
        float v = acc[i][j][r] + bv;
        if (mode == 0) v *= scale;
        if (mode <= 1) {           // [N,H,L,D]
          int l = row >> 1, nb = row & 1, h = col >> 7, d = col & 127;
          outb[(((size_t)(nb * NH + h) * L_SEQ) + l) * DH + d] = (__bf16)v;
        } else if (mode == 2) {    // [N,H,D,L]
          int l = row >> 1, nb = row & 1, h = col >> 7, d = col & 127;
          outb[(((size_t)(nb * NH + h) * DH) + d) * L_SEQ + l] = (__bf16)v;
        } else {                   // fp32 [M,E]
          outf[(size_t)row * EMB + col] = v;
        }
      }
    }
  }
}

// ---------------------------------------------------------------- flash attention
// grid: (L/64 query blocks, N*H). 128 threads = 4 waves; wave w owns query rows
// qb*64 + w*16 .. +15. Online softmax, causal mask matches ref's -10000 fill.
__global__ __launch_bounds__(128)
void attn_wmma(const __bf16* __restrict__ Qg, const __bf16* __restrict__ Kg,
               const __bf16* __restrict__ Vtg, __bf16* __restrict__ ctx) {
  constexpr int QST = DH + 8;   // 136
  constexpr int KST = DH + 8;
  constexpr int VST = 64 + 8;   // 72
  constexpr int PST = 64 + 8;
  __shared__ __bf16 Qs[64 * QST];
  __shared__ __bf16 Ks[64 * KST];
  __shared__ __bf16 Vs[DH * VST];
  __shared__ __bf16 Ps[4 * 16 * PST];

  const int wave = threadIdx.x >> 5;
  const int lane = threadIdx.x & 31;
  const int qb = blockIdx.x;                  // query block 0..31
  const int nh = blockIdx.y;                  // nb*NH + h
  const __bf16* Qh = Qg  + (size_t)nh * L_SEQ * DH;
  const __bf16* Kh = Kg  + (size_t)nh * L_SEQ * DH;
  const __bf16* Vh = Vtg + (size_t)nh * DH * L_SEQ;

  // load Q tile 64x128: 1024 x 16B segments, 8 per thread (async; joined below)
  #pragma unroll
  for (int t = 0; t < 8; ++t) {
    int s  = threadIdx.x + t * 128;
    int r  = s >> 4;
    int cs = (s & 15) * 8;
    copy16_g2l(&Qh[(size_t)(qb * 64 + r) * DH + cs], &Qs[r * QST + cs]);
  }

  v8f o[8] = {};                  // 16 rows x 128 d per wave
  float mrow[8], lrow[8];
  #pragma unroll
  for (int r = 0; r < 8; ++r) { mrow[r] = -1e30f; lrow[r] = 0.f; }

  for (int kb = 0; kb <= qb; ++kb) {
    __syncthreads();
    #pragma unroll
    for (int t = 0; t < 8; ++t) {           // K tile 64x128
      int s  = threadIdx.x + t * 128;
      int r  = s >> 4;
      int cs = (s & 15) * 8;
      copy16_g2l(&Kh[(size_t)(kb * 64 + r) * DH + cs], &Ks[r * KST + cs]);
    }
    #pragma unroll
    for (int t = 0; t < 8; ++t) {           // V^T tile 128x64
      int s  = threadIdx.x + t * 128;
      int r  = s >> 3;
      int cs = (s & 7) * 8;
      copy16_g2l(&Vh[(size_t)r * L_SEQ + kb * 64 + cs], &Vs[r * VST + cs]);
    }
    async_join();
    __syncthreads();

    // S (16x64) = Q_wave (16x128) * K_tile^T
    v8f s4[4] = {};
    #pragma unroll
    for (int kk = 0; kk < DH; kk += 32) {
      v16bf aq = frag_a(&Qs[(wave * 16) * QST + kk], QST);
      #pragma unroll
      for (int j = 0; j < 4; ++j) {
        v16bf bk = frag_b(&Ks[(j * 16) * KST + kk], KST);
        s4[j] = wmma_bf16(aq, bk, s4[j]);
      }
    }

    const int qrow0 = qb * 64 + wave * 16 + ((lane >> 4) << 3);
    if (kb == qb) {                          // causal mask on diagonal block
      #pragma unroll
      for (int j = 0; j < 4; ++j) {
        int kcol = kb * 64 + j * 16 + (lane & 15);
        #pragma unroll
        for (int r = 0; r < 8; ++r)
          if (kcol > qrow0 + r) s4[j][r] = -10000.0f;
      }
    }

    // online softmax: each row lives in 16 lanes of a half-wave
    #pragma unroll
    for (int r = 0; r < 8; ++r) {
      float smax = fmaxf(fmaxf(s4[0][r], s4[1][r]), fmaxf(s4[2][r], s4[3][r]));
      #pragma unroll
      for (int m = 1; m < 16; m <<= 1) smax = fmaxf(smax, __shfl_xor(smax, m, 32));
      float mnew  = fmaxf(mrow[r], smax);
      float alpha = __expf(mrow[r] - mnew);
      mrow[r] = mnew;
      float psum = 0.f;
      #pragma unroll
      for (int j = 0; j < 4; ++j) {
        float p = __expf(s4[j][r] - mnew);
        s4[j][r] = p;
        psum += p;
      }
      #pragma unroll
      for (int m = 1; m < 16; m <<= 1) psum += __shfl_xor(psum, m, 32);
      lrow[r] = lrow[r] * alpha + psum;
      #pragma unroll
      for (int jd = 0; jd < 8; ++jd) o[jd][r] *= alpha;
    }

    // stage P as bf16 in per-wave LDS region (C-layout -> A-layout bounce);
    // same-wave DS ops are in-order, no barrier needed.
    __bf16* Pw = &Ps[wave * 16 * PST];
    {
      const int prow = (lane >> 4) << 3;
      const int pcol = lane & 15;
      #pragma unroll
      for (int j = 0; j < 4; ++j)
        #pragma unroll
        for (int r = 0; r < 8; ++r)
          Pw[(prow + r) * PST + j * 16 + pcol] = (__bf16)s4[j][r];
    }

    // O (16x128) += P (16x64) * V (64x128), B = V^T tile
    #pragma unroll
    for (int kk = 0; kk < 64; kk += 32) {
      v16bf ap = frag_a(&Pw[kk], PST);
      #pragma unroll
      for (int jd = 0; jd < 8; ++jd) {
        v16bf bv = frag_b(&Vs[(jd * 16) * VST + kk], VST);
        o[jd] = wmma_bf16(ap, bv, o[jd]);
      }
    }
  }

  // normalize and store ctx[l*N+nb, h*DH+d] as bf16
  const int nb = nh >> 4, h = nh & 15;
  #pragma unroll
  for (int jd = 0; jd < 8; ++jd) {
    int d = jd * 16 + (lane & 15);
    #pragma unroll
    for (int r = 0; r < 8; ++r) {
      int l = qb * 64 + wave * 16 + ((lane >> 4) << 3) + r;
      float v = o[jd][r] / lrow[r];
      ctx[(size_t)(l * NBATCH + nb) * EMB + h * DH + d] = (__bf16)v;
    }
  }
}

// ---------------------------------------------------------------- launch

extern "C" void kernel_launch(void* const* d_in, const int* in_sizes, int n_in,
                              void* d_out, int out_size, void* d_ws, size_t ws_size,
                              hipStream_t stream) {
  (void)in_sizes; (void)n_in; (void)out_size; (void)ws_size;
  const float* query    = (const float*)d_in[0];
  const float* q_proj   = (const float*)d_in[1];
  const float* q_bias   = (const float*)d_in[2];
  const float* k_proj   = (const float*)d_in[3];
  const float* k_bias   = (const float*)d_in[4];
  const float* v_proj   = (const float*)d_in[5];
  const float* v_bias   = (const float*)d_in[6];
  const float* out_proj = (const float*)d_in[7];
  const float* out_bias = (const float*)d_in[8];
  float* out = (float*)d_out;

  const size_t WN = (size_t)EMB * EMB;      // 4M elements per weight
  const size_t XN = (size_t)MROWS * EMB;    // 8M elements per activation
  __bf16* WQ = (__bf16*)d_ws;
  __bf16* WK = WQ + WN;
  __bf16* WV = WK + WN;
  __bf16* WO = WV + WN;
  __bf16* Xb = WO + WN;
  __bf16* Qh = Xb + XN;
  __bf16* Kh = Qh + XN;
  __bf16* Vt = Kh + XN;
  __bf16* CT = Vt + XN;

  const int wn4 = (int)(WN / 4), xn4 = (int)(XN / 4);
  cvt_f32_bf16<<<wn4 / 256, 256, 0, stream>>>(q_proj,   WQ, wn4);
  cvt_f32_bf16<<<wn4 / 256, 256, 0, stream>>>(k_proj,   WK, wn4);
  cvt_f32_bf16<<<wn4 / 256, 256, 0, stream>>>(v_proj,   WV, wn4);
  cvt_f32_bf16<<<wn4 / 256, 256, 0, stream>>>(out_proj, WO, wn4);
  cvt_f32_bf16<<<xn4 / 256, 256, 0, stream>>>(query,    Xb, xn4);

  dim3 ggrid(EMB / 128, MROWS / 128);
  gemm_wmma<<<ggrid, 128, 0, stream>>>(Xb, WQ, q_bias, Qh, nullptr, 0);
  gemm_wmma<<<ggrid, 128, 0, stream>>>(Xb, WK, k_bias, Kh, nullptr, 1);
  gemm_wmma<<<ggrid, 128, 0, stream>>>(Xb, WV, v_bias, Vt, nullptr, 2);

  attn_wmma<<<dim3(L_SEQ / 64, NBATCH * NH), 128, 0, stream>>>(Qh, Kh, Vt, CT);

  gemm_wmma<<<ggrid, 128, 0, stream>>>(CT, WO, out_bias, nullptr, out, 3);
}
